// feature_matching_costvol_16999480558431
// MI455X (gfx1250) — matrix operand order repaired and checked by reading, not compiled, not run on hardware
//
#include <hip/hip_runtime.h>
#include <hip/hip_bf16.h>
#include <stdint.h>

// CDNA5 / gfx1250: wave32, v_wmma_f32_16x16x32_bf16 (f32 accumulate).
typedef __attribute__((ext_vector_type(16))) __bf16 v16bf;
typedef __attribute__((ext_vector_type(2)))  __bf16 v2bf;
typedef __attribute__((ext_vector_type(8)))  float  v8f;

// Fragment viewed either as the WMMA vector or as 8 packed-pair dwords.
union Frag { v16bf v; uint32_t u[8]; };

// fp32 -> (hi, lo) bf16 split, two elements packed per dword.
static __device__ __forceinline__ void bf16_split2(float x, float y,
                                                   uint32_t& hi, uint32_t& lo) {
    const __bf16 hx = (__bf16)x, hy = (__bf16)y;
    v2bf ph; ph.x = hx; ph.y = hy;
    v2bf pl; pl.x = (__bf16)(x - (float)hx); pl.y = (__bf16)(y - (float)hy);
    hi = __builtin_bit_cast(uint32_t, ph);
    lo = __builtin_bit_cast(uint32_t, pl);
}

// ---------------------------------------------------------------------------
// Pass 1: split depth (the zero-padded operand) into hi/lo bf16 planes with a
// zero halo, packed as channel-pair dwords:
//   plane[b][q][rp][cp] = pack(bf16(src[b][2q][rp-4][cp-4]), bf16(src[b][2q+1][..]))
// Unconditional clamped loads + cndmask select (no exec divergence around vmem).
// ---------------------------------------------------------------------------
template<int C, int H, int W, int NJT>
__global__ __launch_bounds__(256)
void split_pad_bf16(const float* __restrict__ src,
                    uint32_t* __restrict__ hi, uint32_t* __restrict__ lo)
{
    constexpr int  Cp  = C / 2;
    constexpr int  Hp  = H + 8;
    constexpr int  Wp  = 16 * NJT + 16;
    constexpr long TOT = 8L * Cp * Hp * Wp;

    const long idx = (long)blockIdx.x * 256 + threadIdx.x;
    if (idx >= TOT) return;

    const int cp = (int)(idx % Wp);
    long t = idx / Wp;
    const int rp = (int)(t % Hp); t /= Hp;
    const int q  = (int)(t % Cp);
    const int b  = (int)(t / Cp);

    const int  r   = rp - 4, col = cp - 4;
    const bool ok  = ((unsigned)r < (unsigned)H) && ((unsigned)col < (unsigned)W);
    const int  rc  = min(max(r, 0), H - 1);
    const int  cc  = min(max(col, 0), W - 1);

    const long s0 = (((long)b * C + 2 * q) * H + rc) * W + cc;
    float v0 = src[s0];
    float v1 = src[s0 + (long)H * W];
    v0 = ok ? v0 : 0.0f;
    v1 = ok ? v1 : 0.0f;

    uint32_t ph, pl;
    bf16_split2(v0, v1, ph, pl);
    hi[idx] = ph;
    lo[idx] = pl;
}

// ---------------------------------------------------------------------------
// Pass 2: banded-matmul correlation.
// One wave per (batch, i, 16-wide j tile, di group of 3).
// M = j in tile, K = channels, N = absolute padded b columns.
// Two B tiles per di; outputs are diagonals dj = N-M and dj = 16+N-M.
// fp32 recovered via bf16x3: ahi*bhi + alo*bhi + ahi*blo.
// All strides compile-time -> fragment loads are base + immediate offset.
// ---------------------------------------------------------------------------
template<int C, int H, int W, int NJT>
__global__ __launch_bounds__(32)
void corr9_wmma(const float*    __restrict__ Am,
                const uint32_t* __restrict__ Bhi,
                const uint32_t* __restrict__ Blo,
                float* __restrict__ out)
{
    constexpr int  Cp = C / 2;
    constexpr int  Hp = H + 8;
    constexpr int  Wp = 16 * NJT + 16;
    constexpr long HW = (long)H * W;
    constexpr long PS = (long)Hp * Wp;      // dwords per channel-pair slice

    const int lane = threadIdx.x;           // wave32
    const int mn   = lane & 15;             // M (A row) or N (B col / D col)
    const int sel  = lane >> 4;             // which K-half this lane holds
    const int j0   = blockIdx.x << 4;
    const int i    = blockIdx.y;
    const int g    = blockIdx.z % 3;        // di group: di = 3g..3g+2
    const int bat  = blockIdx.z / 3;

    // A lane base (fp32). Clamped column: clamped lanes only feed D rows with
    // j >= W, which are never stored.
    const int jc = min(j0 + mn, W - 1);
    long aoff = (((long)bat * C + sel * 16) * H + i) * W + jc;

    // B lane base (packed bf16 pair planes, padded: row rp = i+di, col cp = j0+nt*16+mn).
    long boff = (((long)bat * Cp + sel * 8) * Hp + (i + g * 3)) * Wp + j0 + mn;

    v8f acc[3][2];
    #pragma unroll
    for (int d = 0; d < 3; ++d) {
        v8f z = {};
        acc[d][0] = z;
        acc[d][1] = z;
    }

    for (int cb = 0; cb < Cp; cb += 16) {   // 16 channel pairs = 32 channels
        // ---- A fragment, contiguous K layout matching B (sel folded into aoff):
        // pair q -> K = sel*16 + 2q, 2q+1
        Frag ahi, alo;
        #pragma unroll
        for (int q = 0; q < 8; ++q) {
            const int  k0 = 2 * q;
            const float v0 = Am[aoff + (long)k0 * HW];
            const float v1 = Am[aoff + (long)(k0 + 1) * HW];
            bf16_split2(v0, v1, ahi.u[q], alo.u[q]);
        }
        if (cb + 16 < Cp)                   // gfx1250 global_prefetch_b8
            __builtin_prefetch(&Am[aoff + 32 * HW], 0, 1);

        #pragma unroll
        for (int d = 0; d < 3; ++d) {
            #pragma unroll
            for (int nt = 0; nt < 2; ++nt) {
                // ---- B fragment: lanes 0-15 hold K=0..15, lanes 16-31 K=16..31
                // (sel folded into boff); pair q -> K = sel*16 + 2q, 2q+1.
                Frag bh, bl;
                #pragma unroll
                for (int q = 0; q < 8; ++q) {
                    const long o = boff + (long)q * PS + d * Wp + nt * 16;
                    bh.u[q] = Bhi[o];
                    bl.u[q] = Blo[o];
                }
                acc[d][nt] = __builtin_amdgcn_wmma_f32_16x16x32_bf16(
                    false, ahi.v, false, bh.v, (short)0, acc[d][nt], false, false);
                acc[d][nt] = __builtin_amdgcn_wmma_f32_16x16x32_bf16(
                    false, alo.v, false, bh.v, (short)0, acc[d][nt], false, false);
                acc[d][nt] = __builtin_amdgcn_wmma_f32_16x16x32_bf16(
                    false, ahi.v, false, bl.v, (short)0, acc[d][nt], false, false);
            }
        }
        aoff += 32 * HW;
        boff += 16 * PS;
    }

    // ---- Scatter band diagonals. D layout: elt r -> row M = r + 8*sel, col N = mn.
    #pragma unroll
    for (int d = 0; d < 3; ++d) {
        const int  di    = g * 3 + d;
        const long oBase = (((long)bat * 9 + di) * 9) * HW + (long)i * W;
        #pragma unroll
        for (int r = 0; r < 8; ++r) {
            const int M = r + 8 * sel;
            const int j = j0 + M;
            if (j >= W) continue;
            const int dj0 = mn - M;          // tile 0 diagonal, dj in [0,8]
            if (dj0 >= 0 && dj0 <= 8)
                out[oBase + (long)dj0 * HW + j] = acc[d][0][r];
            const int dj1 = 16 + mn - M;     // tile 1 diagonal, dj in [1,8]
            if (dj1 <= 8)
                out[oBase + (long)dj1 * HW + j] = acc[d][1][r];
        }
    }
}

// ---------------------------------------------------------------------------
extern "C" void kernel_launch(void* const* d_in, const int* in_sizes, int n_in,
                              void* d_out, int out_size, void* d_ws, size_t ws_size,
                              hipStream_t stream) {
    (void)in_sizes; (void)n_in; (void)out_size; (void)ws_size;

    // setup_inputs dict order is interleaved: rgb0, depth0, rgb1, depth1, ...
    const float* rgb[4];
    const float* dep[4];
    for (int s = 0; s < 4; ++s) {
        rgb[s] = (const float*)d_in[2 * s];
        dep[s] = (const float*)d_in[2 * s + 1];
    }

    // d_out = corr3 | corr2 | corr1 | corr0 (reference returns reversed tuple).
    static const int Hs[4] = {112, 56, 28, 14};
    long off[4];
    long run = 0;
    for (int s = 3; s >= 0; --s) { off[s] = run; run += 8L * 81 * Hs[s] * Hs[s]; }

    uint32_t* ws = (uint32_t*)d_ws;
    long wsoff = 0;   // dwords

#define LAUNCH_SCALE(S, Cc, Hh, NJTn)                                          \
    do {                                                                       \
        constexpr int C_ = Cc, H_ = Hh, NJT_ = NJTn;                           \
        constexpr int Cp_ = C_ / 2, Hp_ = H_ + 8, Wp_ = 16 * NJT_ + 16;        \
        const long planeDw = 8L * Cp_ * Hp_ * Wp_;                             \
        uint32_t* bh = ws + wsoff; wsoff += planeDw;                           \
        uint32_t* bl = ws + wsoff; wsoff += planeDw;                           \
        const unsigned nblk = (unsigned)((planeDw + 255) / 256);               \
        split_pad_bf16<C_, H_, H_, NJT_>                                       \
            <<<dim3(nblk), dim3(256), 0, stream>>>(dep[S], bh, bl);            \
        corr9_wmma<C_, H_, H_, NJT_>                                           \
            <<<dim3(NJT_, H_, 8 * 3), dim3(32), 0, stream>>>(                  \
                rgb[S], bh, bl, (float*)d_out + off[S]);                       \
    } while (0)

    LAUNCH_SCALE(0,  64, 112, 7);
    LAUNCH_SCALE(1, 128,  56, 4);
    LAUNCH_SCALE(2, 256,  28, 2);
    LAUNCH_SCALE(3, 512,  14, 1);
#undef LAUNCH_SCALE
}
